// MultiheadAttention_55482387529933
// MI455X (gfx1250) — matrix operand order repaired and checked by reading, not compiled, not run on hardware
//
#include <hip/hip_runtime.h>
#include <hip/hip_bf16.h>

// MI455X (gfx1250) multi-head attention forward.
// B=2, T=4096, E=512, H=8, D=64. fp32 in/out, f16 WMMA internally (fp32 accum).

typedef _Float16 f16;
typedef __attribute__((ext_vector_type(16))) _Float16 v16h;
typedef __attribute__((ext_vector_type(8)))  _Float16 v8h;
typedef __attribute__((ext_vector_type(8)))  float    v8f;

#define BATCH 2
#define SEQ   4096
#define EMB   512
#define NH    8
#define HD    64

#if __has_builtin(__builtin_amdgcn_exp2f)
#define EXP2F __builtin_amdgcn_exp2f
#else
#define EXP2F exp2f
#endif

__device__ __forceinline__ v8f wmma16(v16h a, v16h b, v8f c) {
  return __builtin_amdgcn_wmma_f32_16x16x32_f16(
      false, a, false, b, (short)0, c, false, false);
}

// Build a 16-half fragment from two 16-byte-aligned 8-half chunks.
__device__ __forceinline__ v16h ld2(const _Float16* p0, const _Float16* p1) {
  v8h a = *(const v8h*)p0;
  v8h b = *(const v8h*)p1;
  v16h r;
#pragma unroll
  for (int i = 0; i < 8; ++i) { r[i] = a[i]; r[i + 8] = b[i]; }
  return r;
}

// CDNA5 async copy: 16B per lane, global -> LDS, tracked by ASYNCcnt.
__device__ __forceinline__ void async_b128(void* lds_ptr, const void* gptr) {
  uint32_t loff = (uint32_t)(size_t)lds_ptr;   // low 32b of generic addr = LDS offset
  uint64_t ga   = (uint64_t)(size_t)gptr;
  asm volatile("global_load_async_to_lds_b128 %0, %1, off"
               :: "v"(loff), "v"(ga) : "memory");
}
__device__ __forceinline__ void wait_async0() {
  asm volatile("s_wait_asynccnt 0" ::: "memory");
}

// ---------------------------------------------------------------------------
// GEMM: Out(M,N) = X(M,K) @ W(N,K)^T + bias(N),  M=B*T=8192, N=K=E=512.
// MODE 0: store f16 [B,H,T,D] (q,k)   MODE 1: store f16 [B,H,D,T] (v^T)
// MODE 2: X is f16 (ctx, async-staged), store f32 [M,E] (final output)
// ---------------------------------------------------------------------------
template <int MODE>
__global__ __launch_bounds__(256) void gemm_wmma(const void* __restrict__ Xv,
                                                 const float* __restrict__ W,
                                                 const float* __restrict__ bias,
                                                 void* __restrict__ Ov) {
  constexpr int K = EMB;
  constexpr int BM = 128, BN = 128, BK = 32;
  constexpr int LDT = BK + 8;  // 40 halves (80B stride, 16B aligned)
  __shared__ _Float16 ldsX[BM * LDT];
  __shared__ _Float16 ldsW[BN * LDT];

  const int tid = threadIdx.x;
  const int w  = tid >> 5;
  const int l  = tid & 31;
  const int hi = l >> 4;
  const int nn = l & 15;
  const int wm = w >> 1;
  const int wn = w & 1;
  const int m0 = blockIdx.y * BM;
  const int n0 = blockIdx.x * BN;

  const v8f vz = {};
  v8f acc[2][4];
#pragma unroll
  for (int mt = 0; mt < 2; ++mt)
#pragma unroll
    for (int nt = 0; nt < 4; ++nt) acc[mt][nt] = vz;

  for (int k0 = 0; k0 < K; k0 += BK) {
    __syncthreads();
    // ---- stage X tile (128 x 32) ----
    if (MODE == 2) {
      const _Float16* X = (const _Float16*)Xv;
#pragma unroll
      for (int it = 0; it < 2; ++it) {
        int chunk = tid + it * 256;          // 512 chunks of 8 halves
        int r = chunk >> 2, c = chunk & 3;
        async_b128(&ldsX[r * LDT + c * 8],
                   X + (size_t)(m0 + r) * K + k0 + c * 8);
      }
    } else {
      const float* X = (const float*)Xv;
#pragma unroll
      for (int it = 0; it < 4; ++it) {
        int chunk = tid + it * 256;          // 1024 chunks of 4 floats
        int r = chunk >> 3, c = chunk & 7;
        float4 d = *(const float4*)(X + (size_t)(m0 + r) * K + k0 + c * 4);
        union { _Float16 h[4]; uint2 u; } cv;
        cv.h[0] = (f16)d.x; cv.h[1] = (f16)d.y; cv.h[2] = (f16)d.z; cv.h[3] = (f16)d.w;
        *(uint2*)&ldsX[r * LDT + c * 4] = cv.u;
      }
    }
    // ---- stage W tile (rows = output features) ----
#pragma unroll
    for (int it = 0; it < 4; ++it) {
      int chunk = tid + it * 256;
      int r = chunk >> 3, c = chunk & 7;
      float4 d = *(const float4*)(W + (size_t)(n0 + r) * K + k0 + c * 4);
      union { _Float16 h[4]; uint2 u; } cv;
      cv.h[0] = (f16)d.x; cv.h[1] = (f16)d.y; cv.h[2] = (f16)d.z; cv.h[3] = (f16)d.w;
      *(uint2*)&ldsW[r * LDT + c * 4] = cv.u;
    }
    if (MODE == 2) wait_async0();
    __syncthreads();

    v16h aF[2];
#pragma unroll
    for (int mt = 0; mt < 2; ++mt) {
      const _Float16* p = &ldsX[(wm * 32 + mt * 16 + nn) * LDT + hi * 8];
      aF[mt] = ld2(p, p + 16);
    }
    v16h bF[4];
#pragma unroll
    for (int nt = 0; nt < 4; ++nt) {
      const _Float16* p = &ldsW[(wn * 64 + nt * 16 + nn) * LDT + hi * 8];
      bF[nt] = ld2(p, p + 16);
    }
#pragma unroll
    for (int mt = 0; mt < 2; ++mt)
#pragma unroll
      for (int nt = 0; nt < 4; ++nt)
        acc[mt][nt] = wmma16(aF[mt], bF[nt], acc[mt][nt]);
  }

#pragma unroll
  for (int mt = 0; mt < 2; ++mt)
#pragma unroll
    for (int nt = 0; nt < 4; ++nt)
#pragma unroll
      for (int r = 0; r < 8; ++r) {
        int gm = m0 + wm * 32 + mt * 16 + r + 8 * hi;
        int gn = n0 + wn * 64 + nt * 16 + nn;
        float v = acc[mt][nt][r] + bias[gn];
        if (MODE == 2) {
          ((float*)Ov)[(size_t)gm * EMB + gn] = v;
        } else {
          int b = gm >> 12, t = gm & (SEQ - 1);
          int h = gn >> 6,  d = gn & (HD - 1);
          if (MODE == 0)
            ((_Float16*)Ov)[(((size_t)b * NH + h) * SEQ + t) * HD + d] = (f16)v;
          else
            ((_Float16*)Ov)[(((size_t)b * NH + h) * HD + d) * SEQ + t] = (f16)v;
        }
      }
}

// ---------------------------------------------------------------------------
// Flash attention: grid (B*H, T/128). 8 waves x 16 query rows; 64-key tiles,
// double-buffered async global->LDS staging, exp2-domain online softmax,
// row-sums via WMMA against a ones matrix.
// ---------------------------------------------------------------------------
__global__ __launch_bounds__(256) void attn_wmma(const _Float16* __restrict__ q16,
                                                 const _Float16* __restrict__ k16,
                                                 const _Float16* __restrict__ vT16,
                                                 _Float16* __restrict__ ctx16) {
  constexpr int TK = 64;
  constexpr int LK = TK + 8;                  // 72-half row stride
  __shared__ _Float16 ldsK[2][TK * LK];       // K tile   [j][d]
  __shared__ _Float16 ldsV[2][HD * LK];       // V^T tile [d][j]
  __shared__ _Float16 ldsP[8][16 * LK];       // per-wave P staging (16 x 64)

  const int tid = threadIdx.x;
  const int w = tid >> 5, l = tid & 31, hi = l >> 4, nn = l & 15;
  const int bh = blockIdx.x;
  const int q0 = blockIdx.y * 128 + w * 16;

  // Q fragments with softmax scale folded in (exp2 domain): s*log2(e)/sqrt(D)
  const float SC = 0.125f * 1.44269504f;
  const _Float16* qrow = q16 + ((size_t)bh * SEQ + q0 + nn) * HD;
  v16h qF[2];
#pragma unroll
  for (int kc = 0; kc < 2; ++kc) {
    const _Float16* p = qrow + kc * 32 + hi * 8;
    qF[kc] = ld2(p, p + 16);
#pragma unroll
    for (int i = 0; i < 16; ++i) qF[kc][i] = (f16)((float)qF[kc][i] * SC);
  }

  v16h ones;
#pragma unroll
  for (int i = 0; i < 16; ++i) ones[i] = (f16)1.0f;

  const v8f vz = {};
  v8f ctx[4];
#pragma unroll
  for (int dt = 0; dt < 4; ++dt) ctx[dt] = vz;
  float m_run[8], l_run[8];
#pragma unroll
  for (int r = 0; r < 8; ++r) { m_run[r] = -1e30f; l_run[r] = 0.0f; }

  // async tile staging: 64 rows x 64 halves each for K and V^T
  const _Float16* kBase = k16 + (size_t)bh * SEQ * HD;
  const _Float16* vBase = vT16 + (size_t)bh * HD * SEQ;
  auto stage = [&](int j0, int b) {
#pragma unroll
    for (int it = 0; it < 2; ++it) {
      int idx = tid + it * 256;              // 512 chunks of 8 halves
      int r = idx >> 3, c = idx & 7;
      async_b128(&ldsK[b][r * LK + c * 8], kBase + (size_t)(j0 + r) * HD + c * 8);
      async_b128(&ldsV[b][r * LK + c * 8], vBase + (size_t)r * SEQ + j0 + c * 8);
    }
  };

  stage(0, 0);
  int buf = 0;
  for (int j0 = 0; j0 < SEQ; j0 += TK) {
    wait_async0();
    __syncthreads();
    if (j0 + TK < SEQ) stage(j0 + TK, buf ^ 1);

    // scores: 4 key N-tiles x (D=64 -> 2 chained WMMAs)
    v8f s[4];
#pragma unroll
    for (int nt = 0; nt < 4; ++nt) {
      v8f t = vz;
#pragma unroll
      for (int kc = 0; kc < 2; ++kc) {
        const _Float16* p = &ldsK[buf][(nt * 16 + nn) * LK + kc * 32 + hi * 8];
        t = wmma16(qF[kc], ld2(p, p + 16), t);
      }
      s[nt] = t;
    }

    // online softmax (exp2 domain); only the MAX needs a lane reduction
#pragma unroll
    for (int r = 0; r < 8; ++r) {
      float v0 = s[0][r], v1 = s[1][r], v2 = s[2][r], v3 = s[3][r];
      float mx = fmaxf(fmaxf(v0, v1), fmaxf(v2, v3));
#pragma unroll
      for (int m = 1; m < 16; m <<= 1) mx = fmaxf(mx, __shfl_xor(mx, m, 32));
      float mnew  = fmaxf(m_run[r], mx);
      float alpha = EXP2F(m_run[r] - mnew);
      m_run[r] = mnew;
      l_run[r] *= alpha;
#pragma unroll
      for (int dt = 0; dt < 4; ++dt) ctx[dt][r] *= alpha;
      float p0 = EXP2F(v0 - mnew);
      float p1 = EXP2F(v1 - mnew);
      float p2 = EXP2F(v2 - mnew);
      float p3 = EXP2F(v3 - mnew);
      int row = (r + 8 * hi) * LK;
      ldsP[w][row + nn]      = (f16)p0;
      ldsP[w][row + 16 + nn] = (f16)p1;
      ldsP[w][row + 32 + nn] = (f16)p2;
      ldsP[w][row + 48 + nn] = (f16)p3;
    }

    // P fragments (16 x 64)
    v16h pF[2];
#pragma unroll
    for (int kc = 0; kc < 2; ++kc) {
      const _Float16* p = &ldsP[w][nn * LK + kc * 32 + hi * 8];
      pF[kc] = ld2(p, p + 16);
    }
    // row sums via matrix core: rs = P @ ones
    v8f rs = wmma16(pF[0], ones, vz);
    rs = wmma16(pF[1], ones, rs);
    // ctx += P @ V
#pragma unroll
    for (int dt = 0; dt < 4; ++dt) {
#pragma unroll
      for (int kc = 0; kc < 2; ++kc) {
        const _Float16* vp = &ldsV[buf][(dt * 16 + nn) * LK + kc * 32 + hi * 8];
        ctx[dt] = wmma16(pF[kc], ld2(vp, vp + 16), ctx[dt]);
      }
    }
#pragma unroll
    for (int r = 0; r < 8; ++r) l_run[r] += rs[r];
    buf ^= 1;
  }

  const int b = bh / NH, h = bh % NH;
#pragma unroll
  for (int r = 0; r < 8; ++r) {
    float inv = 1.0f / l_run[r];
    int t = q0 + r + 8 * hi;
#pragma unroll
    for (int dt = 0; dt < 4; ++dt) {
      float v = ctx[dt][r] * inv;
      ctx16[((size_t)b * SEQ + t) * EMB + h * HD + dt * 16 + nn] = (f16)v;
    }
  }
}

// ---------------------------------------------------------------------------
extern "C" void kernel_launch(void* const* d_in, const int* in_sizes, int n_in,
                              void* d_out, int out_size, void* d_ws, size_t ws_size,
                              hipStream_t stream) {
  (void)in_sizes; (void)n_in; (void)out_size; (void)ws_size;
  const float* query = (const float*)d_in[0];
  const float* key_  = (const float*)d_in[1];
  const float* value = (const float*)d_in[2];
  const float* Wq = (const float*)d_in[3];
  const float* bq = (const float*)d_in[4];
  const float* Wk = (const float*)d_in[5];
  const float* bk = (const float*)d_in[6];
  const float* Wv = (const float*)d_in[7];
  const float* bv = (const float*)d_in[8];
  const float* Wo = (const float*)d_in[9];
  const float* bo = (const float*)d_in[10];
  float* out = (float*)d_out;

  const size_t nEl = (size_t)BATCH * SEQ * EMB;   // 4 Mi elements
  _Float16* q16 = (_Float16*)d_ws;
  _Float16* k16 = q16 + nEl;
  _Float16* vT  = k16 + nEl;
  _Float16* cx  = vT + nEl;

  dim3 gGemm(EMB / 128, (BATCH * SEQ) / 128);   // (4, 64)
  dim3 gAttn(BATCH * NH, SEQ / 128);            // (16, 32)

  gemm_wmma<0><<<gGemm, 256, 0, stream>>>(query, Wq, bq, q16);
  gemm_wmma<0><<<gGemm, 256, 0, stream>>>(key_,  Wk, bk, k16);
  gemm_wmma<1><<<gGemm, 256, 0, stream>>>(value, Wv, bv, vT);
  attn_wmma<<<gAttn, 256, 0, stream>>>(q16, k16, vT, cx);
  gemm_wmma<2><<<gGemm, 256, 0, stream>>>(cx, Wo, bo, out);
}
